// GaussianConv_34179349742144
// MI455X (gfx1250) — matrix operand order repaired and checked by reading, not compiled, not run on hardware
//
#include <hip/hip_runtime.h>
#include <hip/hip_bf16.h>

// ---------------------------------------------------------------------------
// GaussianConv pyramid on MI455X (gfx1250, wave32, WMMA, LDS-staged gathers).
// All gathered GEMMs run on v_wmma_f32_16x16x32_f16 with f32 accumulators.
// ---------------------------------------------------------------------------

typedef __attribute__((ext_vector_type(16))) _Float16 v16h;
typedef __attribute__((ext_vector_type(8)))  _Float16 v8h;
typedef __attribute__((ext_vector_type(8)))  float    v8f;

#define NPTS  100000
#define NSEG  12500
#define KNN   8

// ---------------- conversion kernels ----------------

__global__ void cvt_f32_to_f16(_Float16* __restrict__ dst,
                               const float* __restrict__ src, int n) {
  int i = blockIdx.x * blockDim.x + threadIdx.x;
  if (i < n) dst[i] = (_Float16)src[i];
}

// weight convert with optional zero row padding (for Cout=3 -> 16)
__global__ void cvt_wpad(_Float16* __restrict__ dst, const float* __restrict__ src,
                         int srcRows, int cols, int totalDst) {
  int i = blockIdx.x * blockDim.x + threadIdx.x;
  if (i >= totalDst) return;
  int r = i / cols;
  dst[i] = (r < srcRows) ? (_Float16)src[i] : (_Float16)0.0f;
}

// ---------------- segment mean (downsample) ----------------

__global__ void seg_scatter(float* __restrict__ sums, float* __restrict__ cnt,
                            const _Float16* __restrict__ f,
                            const int* __restrict__ labels, int npts, int ch) {
  int i = blockIdx.x * blockDim.x + threadIdx.x;
  if (i >= npts * ch) return;
  int p = i / ch;
  int c = i - p * ch;
  int s = labels[p];
  atomicAdd(&sums[(size_t)s * ch + c], (float)f[i]);
  if (c == 0) atomicAdd(&cnt[s], 1.0f);
}

__global__ void seg_finalize(_Float16* __restrict__ out, const float* __restrict__ sums,
                             const float* __restrict__ cnt, int nseg, int ch) {
  int i = blockIdx.x * blockDim.x + threadIdx.x;
  if (i >= nseg * ch) return;
  int s = i / ch;
  float d = cnt[s];
  out[i] = (_Float16)(sums[i] / fmaxf(d, 1.0f));
}

// ---------------- big fused gather + GEMM (LDS-staged A, WMMA) -------------
//
// out[m, co] = act( sum_{k<8, c<CIN} feat[src(m,k), c] * W[co, k*CIN + c] + b[co] )
//
// Block = 4 waves, 64 output rows (4 M-tiles), full COUT columns.
// Each 32-wide K chunk of the gathered A matrix (64x32 f16, 4KB) is staged
// once into double-buffered LDS; every wave computes all 4 M-tiles for its
// COUT/64 column tiles, reusing each register-resident B fragment across
// 4 WMMAs. LDS A rows use a 40-half stride so 16B fragment reads are
// bank-conflict-free.
template<int CIN, int COUT, bool ACT, bool LBL>
__launch_bounds__(128)
__global__ void gconv_big(const _Float16* __restrict__ feat,
                          const int* __restrict__ idx,
                          const int* __restrict__ labels,
                          const _Float16* __restrict__ w,     // [COUT, 8*CIN] f16
                          const float* __restrict__ bias,     // [COUT] f32
                          _Float16* __restrict__ outh,        // [M, COUT] f16
                          int M) {
  constexpr int KT  = 8 * CIN;
  constexpr int NQ  = KT / 32;          // number of 32-wide K chunks
  constexpr int TPW = COUT / 16 / 4;    // column tiles per wave
  constexpr int AS  = 40;               // padded LDS row stride (halves)

  __shared__ __align__(128) _Float16 As[2][64 * AS];
  __shared__ int srcs[64 * KNN];

  const int tid  = threadIdx.x;
  const int lane = tid & 31;
  const int wave = tid >> 5;
  const int n    = lane & 15;   // A row within tile / B,C,D column
  const int h    = lane >> 4;   // half-wave group
  const int m0   = blockIdx.x * 64;

  // resolve the 64 rows x 8 neighbors gather table once per block
#pragma unroll
  for (int j = 0; j < 4; ++j) {
    int e = tid * 4 + j;                // 512 entries
    int r = e >> 3, k = e & 7;
    int row = m0 + r;
    int msafe = (row < M) ? row : 0;
    int s = idx[msafe * KNN + k];
    if (LBL) s = labels[s];
    srcs[e] = s;
  }
  __syncthreads();

  // stage one 64x32 gathered A chunk into LDS (each thread copies 32B)
  auto stage = [&](int qi, int buf) {
    const int q = qi * 32;
    const int k = q / CIN;              // CIN is a power of two -> shifts
    const int c = q - k * CIN;
    const int r = tid >> 1, part = tid & 1;
    const _Float16* gp = feat + (size_t)srcs[r * KNN + k] * CIN + c + part * 16;
    v8h x0 = *reinterpret_cast<const v8h*>(gp);
    v8h x1 = *reinterpret_cast<const v8h*>(gp + 8);
    *reinterpret_cast<v8h*>(&As[buf][r * AS + part * 16])     = x0;
    *reinterpret_cast<v8h*>(&As[buf][r * AS + part * 16 + 8]) = x1;
  };

  v8f acc[4][TPW];
#pragma unroll
  for (int mt = 0; mt < 4; ++mt)
#pragma unroll
    for (int t = 0; t < TPW; ++t) acc[mt][t] = (v8f)(0.0f);

  stage(0, 0);
  __syncthreads();

  for (int qi = 0; qi < NQ; ++qi) {
    const int buf = qi & 1;
    if (qi + 1 < NQ) stage(qi + 1, buf ^ 1);   // prefetch next chunk

    // A fragments for the 4 M-tiles from LDS
    // (contiguous k order chosen here for A and B: lane group h holds K in [16h,16h+16))
    v16h a[4];
#pragma unroll
    for (int mt = 0; mt < 4; ++mt) {
      const _Float16* ap = &As[buf][(mt * 16 + n) * AS + h * 16];
      v8h lo = *reinterpret_cast<const v8h*>(ap);
      v8h hi = *reinterpret_cast<const v8h*>(ap + 8);
#pragma unroll
      for (int i = 0; i < 8; ++i) { a[mt][i] = lo[i]; a[mt][i + 8] = hi[i]; }
    }

#pragma unroll
    for (int t = 0; t < TPW; ++t) {
      const int co = (t * 4 + wave) * 16;      // wave-interleaved column tiles
      // B fragment: lane n = output channel, lane group h holds K in [16h,16h+16)
      const _Float16* wp = w + (size_t)(co + n) * KT + qi * 32 + h * 16;
      v16h b = *reinterpret_cast<const v16h*>(wp);
#pragma unroll
      for (int mt = 0; mt < 4; ++mt)
        acc[mt][t] = __builtin_amdgcn_wmma_f32_16x16x32_f16(
            false, a[mt], false, b, (short)0, acc[mt][t], false, false);
    }
    __syncthreads();
  }

  // epilogue: D layout — lane n = column, VGPR r = row h*8 + r
#pragma unroll
  for (int t = 0; t < TPW; ++t) {
    const int col = (t * 4 + wave) * 16 + n;
    const float bv = bias[col];
#pragma unroll
    for (int mt = 0; mt < 4; ++mt) {
#pragma unroll
      for (int r = 0; r < 8; ++r) {
        int row = m0 + mt * 16 + h * 8 + r;
        if (row < M) {
          float v = acc[mt][t][r] + bv;
          if (ACT) v = 1.0f / (1.0f + __expf(-v));
          outh[(size_t)row * COUT + col] = (_Float16)v;
        }
      }
    }
  }
}

// ---------------- small fused gather + GEMM (per-wave, WMMA) ---------------
// Used for the narrow tail layers (Cout 32 / 3) where the big tile is waste.
template<int CIN, int COUT, int NWAVES, bool ACT, bool LBL, bool FINAL>
__launch_bounds__(NWAVES * 32)
__global__ void gconv_wmma(const _Float16* __restrict__ feat,
                           const int* __restrict__ idx,
                           const int* __restrict__ labels,
                           const _Float16* __restrict__ w,
                           const float* __restrict__ bias,
                           _Float16* __restrict__ outh,
                           float* __restrict__ outf,
                           int M, int biasN, int outReal) {
  constexpr int KT  = 8 * CIN;
  constexpr int TPW = COUT / 16 / NWAVES;

  const int lane = threadIdx.x & 31;
  const int wave = threadIdx.x >> 5;
  const int n    = lane & 15;
  const int h    = lane >> 4;
  const int m16  = blockIdx.x * 16;

  const int m     = m16 + n;
  const int msafe = (m < M) ? m : 0;
  int src[KNN];
#pragma unroll
  for (int k = 0; k < KNN; ++k) {
    int s = idx[msafe * KNN + k];
    if (LBL) s = labels[s];
    src[k] = s;
  }

  v8f acc[TPW];
#pragma unroll
  for (int t = 0; t < TPW; ++t) acc[t] = (v8f)(0.0f);

  for (int q = 0; q < KT; q += 32) {
    const int k = q / CIN;
    const int c = q - k * CIN;
    const _Float16* rp = feat + (size_t)src[k] * CIN + c;
    v8h lo = *reinterpret_cast<const v8h*>(rp + h * 16);
    v8h hi = *reinterpret_cast<const v8h*>(rp + h * 16 + 8);
    v16h a;
#pragma unroll
    for (int i = 0; i < 8; ++i) { a[i] = lo[i]; a[i + 8] = hi[i]; }

#pragma unroll
    for (int t = 0; t < TPW; ++t) {
      const int co = (wave * TPW + t) * 16;
      const _Float16* wp = w + (size_t)(co + n) * KT + q + h * 16;
      v16h b = *reinterpret_cast<const v16h*>(wp);
      acc[t] = __builtin_amdgcn_wmma_f32_16x16x32_f16(
          false, a, false, b, (short)0, acc[t], false, false);
    }
  }

#pragma unroll
  for (int t = 0; t < TPW; ++t) {
    const int co = (wave * TPW + t) * 16;
    const int col = co + n;
    const float bv = (col < biasN) ? bias[col] : 0.0f;
#pragma unroll
    for (int r = 0; r < 8; ++r) {
      const int row = m16 + h * 8 + r;
      if (row < M) {
        float v = acc[t][r] + bv;
        if (ACT) v = 1.0f / (1.0f + __expf(-v));
        if (FINAL) {
          if (col < outReal) outf[(size_t)row * outReal + col] = v;
        } else {
          outh[(size_t)row * COUT + col] = (_Float16)v;
        }
      }
    }
  }
}

// ---------------------------------------------------------------------------

extern "C" void kernel_launch(void* const* d_in, const int* in_sizes, int n_in,
                              void* d_out, int out_size, void* d_ws, size_t ws_size,
                              hipStream_t stream) {
  (void)in_sizes; (void)n_in; (void)out_size; (void)ws_size;

  const float* features = (const float*)d_in[0];
  const int*   idx0     = (const int*)d_in[1];
  const int*   idx1     = (const int*)d_in[2];
  const int*   labels   = (const int*)d_in[3];
  const float* k0 = (const float*)d_in[4];   const float* b0 = (const float*)d_in[5];
  const float* k1 = (const float*)d_in[6];   const float* b1 = (const float*)d_in[7];
  const float* k2 = (const float*)d_in[8];   const float* b2 = (const float*)d_in[9];
  const float* k3 = (const float*)d_in[10];  const float* b3 = (const float*)d_in[11];
  const float* k4 = (const float*)d_in[12];  const float* b4 = (const float*)d_in[13];
  float* out = (float*)d_out;

  // ---- workspace carve-up (256B aligned) ----
  char* ws = (char*)d_ws;
  size_t off = 0;
  auto alloc = [&](size_t bytes) -> char* {
    char* p = ws + off;
    off = (off + bytes + 255) & ~(size_t)255;
    return p;
  };
  _Float16* feat16 = (_Float16*)alloc((size_t)NPTS * 256 * 2);
  _Float16* f0h    = (_Float16*)alloc((size_t)NPTS * 256 * 2);
  float*    fmf    = (float*)   alloc((size_t)NSEG * 256 * 4);
  float*    cnt    = (float*)   alloc((size_t)NSEG * 4);
  _Float16* fmh    = (_Float16*)alloc((size_t)NSEG * 256 * 2);
  _Float16* f1h    = (_Float16*)alloc((size_t)NSEG * 128 * 2);
  _Float16* f2h    = (_Float16*)alloc((size_t)NSEG * 64 * 2);
  _Float16* f3h    = (_Float16*)alloc((size_t)NPTS * 32 * 2);
  _Float16* w0h    = (_Float16*)alloc((size_t)256 * 2048 * 2);
  _Float16* w1h    = (_Float16*)alloc((size_t)128 * 2048 * 2);
  _Float16* w2h    = (_Float16*)alloc((size_t)64 * 1024 * 2);
  _Float16* w3h    = (_Float16*)alloc((size_t)32 * 512 * 2);
  _Float16* w4h    = (_Float16*)alloc((size_t)16 * 256 * 2);   // padded 3 -> 16 rows

  const int T = 256;
  auto g = [&](int n) { return (n + T - 1) / T; };

  // ---- f32 -> f16 conversions ----
  cvt_f32_to_f16<<<g(NPTS * 256), T, 0, stream>>>(feat16, features, NPTS * 256);
  cvt_wpad<<<g(256 * 2048), T, 0, stream>>>(w0h, k0, 256, 2048, 256 * 2048);
  cvt_wpad<<<g(128 * 2048), T, 0, stream>>>(w1h, k1, 128, 2048, 128 * 2048);
  cvt_wpad<<<g(64 * 1024), T, 0, stream>>>(w2h, k2, 64, 1024, 64 * 1024);
  cvt_wpad<<<g(32 * 512), T, 0, stream>>>(w3h, k3, 32, 512, 32 * 512);
  cvt_wpad<<<g(16 * 256), T, 0, stream>>>(w4h, k4, 3, 256, 16 * 256);

  const int gbBigFull   = (NPTS + 63) / 64;   // 1563
  const int gbBigCoarse = (NSEG + 63) / 64;   // 196
  const int gbFull      = (NPTS + 15) / 16;   // 6250

  // ---- layer 0: [N,256] -> [N,256], sigmoid ----
  gconv_big<256, 256, true, false><<<gbBigFull, 128, 0, stream>>>(
      feat16, idx0, nullptr, w0h, b0, f0h, NPTS);

  // ---- downsample: segment mean over labels ----
  hipMemsetAsync(fmf, 0, (size_t)NSEG * 256 * 4, stream);
  hipMemsetAsync(cnt, 0, (size_t)NSEG * 4, stream);
  seg_scatter<<<g(NPTS * 256), T, 0, stream>>>(fmf, cnt, f0h, labels, NPTS, 256);
  seg_finalize<<<g(NSEG * 256), T, 0, stream>>>(fmh, fmf, cnt, NSEG, 256);

  // ---- layer 1: [N1,256] -> [N1,128], sigmoid ----
  gconv_big<256, 128, true, false><<<gbBigCoarse, 128, 0, stream>>>(
      fmh, idx1, nullptr, w1h, b1, f1h, NSEG);

  // ---- layer 2: [N1,128] -> [N1,64], sigmoid ----
  gconv_big<128, 64, true, false><<<gbBigCoarse, 128, 0, stream>>>(
      f1h, idx1, nullptr, w2h, b2, f2h, NSEG);

  // ---- layer 3: upsample (labels) fused into gather; [N,64] -> [N,32] ----
  gconv_wmma<64, 32, 2, true, true, false><<<gbFull, 64, 0, stream>>>(
      f2h, idx0, labels, w3h, b3, f3h, nullptr, NPTS, 32, 32);

  // ---- layer 4: [N,32] -> [N,3], no activation, f32 output ----
  gconv_wmma<32, 16, 1, false, false, true><<<gbFull, 32, 0, stream>>>(
      f3h, idx0, nullptr, w4h, b4, nullptr, out, NPTS, 3, 3);
}
